// DNCLSTMNet_11630771438088
// MI455X (gfx1250) — matrix operand order repaired and checked by reading, not compiled, not run on hardware
//
#include <hip/hip_runtime.h>
#include <hip/hip_bf16.h>

// ---------------------------------------------------------------------------
// DNC-LSTM forward for MI455X (gfx1250, wave32, WMMA bf16 + TDM async tensor)
// ---------------------------------------------------------------------------

typedef __attribute__((ext_vector_type(16))) __bf16 v16bf;
typedef __attribute__((ext_vector_type(8)))  __bf16 v8bf;
typedef __attribute__((ext_vector_type(8)))  float  v8f;

#define T_LEN 64
#define BSZ   32
#define EMB   512
#define HIDD  1024
#define OUTD  16000
#define HDN   4
#define NSLOT 128
#define WDIM  64
#define KG    1792   // 512 (emb) + 256 (rv) + 1024 (h)
#define KY    1280   // 1024 (h) + 256 (rv)
#define G4    4096   // 4*HIDD
#define EPSV  1e-6f

#ifndef __has_builtin
#define __has_builtin(x) 0
#endif
#if __has_builtin(__builtin_amdgcn_tensor_load_to_lds) && \
    __has_builtin(__builtin_amdgcn_s_wait_tensorcnt)
#define USE_TDM 1
#else
#define USE_TDM 0
#endif

__device__ __forceinline__ float sigm(float x) { return 1.0f / (1.0f + expf(-x)); }
__device__ __forceinline__ float oneplus_f(float x) {
  float sp = (x > 20.0f) ? x : log1pf(expf(x));
  return 1.0f + sp;
}

#if USE_TDM
typedef __attribute__((ext_vector_type(4))) unsigned int u32x4;
typedef __attribute__((ext_vector_type(8))) int i32x8;
typedef __attribute__((ext_vector_type(4))) int i32x4;

// Issue one TDM 2D tile load: tileW x tileH elements of 2 bytes, row stride
// `strideElems`, from global `gaddr` into LDS byte offset `lds_off`.
// D# layout per CDNA5 ISA ch.8 (group0 128b, group1 256b; groups 2/3 and the
// trailing descriptor group are zero for a plain 2D tile).
__device__ __forceinline__ void tdm_load_2d(unsigned long gaddr, unsigned lds_off,
                                            unsigned tileW, unsigned tileH,
                                            unsigned strideElems) {
  u32x4 g0;
  g0[0] = 1u;                                          // count=1, user mode
  g0[1] = lds_off;                                     // lds_addr (bytes)
  g0[2] = (unsigned)(gaddr & 0xFFFFFFFFu);             // global_addr[31:0]
  g0[3] = (unsigned)((gaddr >> 32) & 0x1FFFFFFu)       // global_addr[56:32]
        | (2u << 30);                                  // type = 2 ("image")
  i32x8 g1;
  g1[0] = (1 << 16);                                   // data_size=1 -> 2 bytes
  g1[1] = (int)((tileW & 0xFFFFu) << 16);              // tensor_dim0[15:0]
  g1[2] = (int)((tileW >> 16) | ((tileH & 0xFFFFu) << 16));  // dim0 hi | dim1 lo
  g1[3] = (int)((tileW & 0xFFFFu) << 16);              // tile_dim0 (dim1 hi = 0)
  g1[4] = (int)(tileH & 0xFFFFu);                      // tile_dim1 (tile_dim2=0)
  g1[5] = (int)strideElems;                            // tensor_dim0_stride lo
  g1[6] = 0;
  g1[7] = 0;
  i32x4 z4 = {0, 0, 0, 0};
  i32x8 z8 = {0, 0, 0, 0, 0, 0, 0, 0};
  __builtin_amdgcn_tensor_load_to_lds(g0, g1, z4, z4, z8, 0);
}
#endif

// ---------------------------------------------------------------------------
// Weight conversion: build bf16 packed weight matrices once per launch.
//   Wfull[4096][1792] = [W_ih (768 cols) | W_hh (1024 cols)]
//   Wy   [16000][1280] = [Why_w (1024 cols) | Wry_w (256 cols)]
//   bias_sum[4096] = b_ih + b_hh
// ---------------------------------------------------------------------------
__global__ void k_convert(const float* __restrict__ W_ih, const float* __restrict__ W_hh,
                          const float* __restrict__ b_ih, const float* __restrict__ b_hh,
                          const float* __restrict__ Why,  const float* __restrict__ Wry,
                          __bf16* __restrict__ Wfull, __bf16* __restrict__ Wy,
                          float* __restrict__ bias_sum) {
  const long nW = (long)G4 * KG;
  const long nY = (long)OUTD * KY;
  const long nB = G4;
  const long total = nW + nY + nB;
  const long stride = (long)gridDim.x * blockDim.x;
  for (long idx = (long)blockIdx.x * blockDim.x + threadIdx.x; idx < total; idx += stride) {
    if (idx < nW) {
      long n = idx / KG, k = idx % KG;
      float v = (k < 768) ? W_ih[n * 768 + k] : W_hh[n * 1024 + (k - 768)];
      Wfull[idx] = (__bf16)v;
    } else if (idx < nW + nY) {
      long r = idx - nW;
      long n = r / KY, k = r % KY;
      float v = (k < 1024) ? Why[n * 1024 + k] : Wry[n * 256 + (k - 1024)];
      Wy[r] = (__bf16)v;
    } else {
      long n = idx - nW - nY;
      bias_sum[n] = b_ih[n] + b_hh[n];
    }
  }
}

// ---------------------------------------------------------------------------
// State init: zero carries, u = EPS, pack x_bf for t=0 = [emb(src[0,b]) | 0 | 0]
// ---------------------------------------------------------------------------
__global__ void k_init(const int* __restrict__ src, const float* __restrict__ emb,
                       float* s_g, float* u_g, float* p_g, float* ww_g,
                       float* rw_g, float* mem_g, float* links_g, __bf16* xb) {
  const long nS  = (long)BSZ * HIDD;
  const long nU  = (long)BSZ * NSLOT;
  const long nRW = (long)BSZ * NSLOT * HDN;
  const long nM  = (long)BSZ * NSLOT * WDIM;
  const long nL  = (long)BSZ * NSLOT * NSLOT;
  const long nX  = (long)BSZ * KG;
  const long total = nS + 3 * nU + nRW + nM + nL + nX;
  const long stride = (long)gridDim.x * blockDim.x;
  for (long idx = (long)blockIdx.x * blockDim.x + threadIdx.x; idx < total; idx += stride) {
    long r = idx;
    if (r < nS)  { s_g[r] = 0.0f; continue; }  r -= nS;
    if (r < nU)  { u_g[r] = EPSV; continue; }  r -= nU;
    if (r < nU)  { p_g[r] = 0.0f; continue; }  r -= nU;
    if (r < nU)  { ww_g[r] = 0.0f; continue; } r -= nU;
    if (r < nRW) { rw_g[r] = 0.0f; continue; } r -= nRW;
    if (r < nM)  { mem_g[r] = 0.0f; continue; } r -= nM;
    if (r < nL)  { links_g[r] = 0.0f; continue; } r -= nL;
    {
      long b = r / KG, k = r % KG;
      float v = 0.0f;
      if (k < EMB) v = emb[(long)src[b] * EMB + k];   // src[t=0][b]
      xb[r] = (__bf16)v;
    }
  }
}

// ---------------------------------------------------------------------------
// Gates GEMM: gates[32][4096] = x_bf[32][1792] * Wfull^T + (b_ih+b_hh)
// 256 threads = 8 waves arranged 2(M)x4(N); wave tile 16x16; grid = 64 WGs.
// ---------------------------------------------------------------------------
__global__ void __launch_bounds__(256)
k_gates(const __bf16* __restrict__ xb, const __bf16* __restrict__ Wfull,
        const float* __restrict__ bias, float* __restrict__ gates) {
  const int wave = threadIdx.x >> 5;
  const int lane = threadIdx.x & 31;
  const int hf = lane >> 4;          // which 16-lane half
  const int lanelo = lane & 15;
  const int m0 = (wave & 1) * 16;
  const int n0 = blockIdx.x * 64 + (wave >> 1) * 16;
  const int m = m0 + lanelo;
  const int n = n0 + lanelo;
  const __bf16* arow = xb + (long)m * KG;
  const __bf16* brow = Wfull + (long)n * KG;

  v8f acc = {0.f, 0.f, 0.f, 0.f, 0.f, 0.f, 0.f, 0.f};
  for (int k0 = 0; k0 < KG; k0 += 32) {
    v8bf a0 = *(const v8bf*)(arow + k0 + hf * 16);
    v8bf a1 = *(const v8bf*)(arow + k0 + hf * 16 + 8);
    v16bf av, bv;
    bv = *(const v16bf*)(brow + k0 + hf * 16);
#pragma unroll
    for (int j = 0; j < 8; ++j) { av[j] = a0[j]; av[j + 8] = a1[j]; }
    acc = __builtin_amdgcn_wmma_f32_16x16x32_bf16(false, av, false, bv,
                                                  (short)0, acc, false, false);
  }
  const float bb = bias[n];
#pragma unroll
  for (int r = 0; r < 8; ++r) {
    int mm = m0 + r + 8 * hf;
    gates[(long)mm * G4 + n] = acc[r] + bb;
  }
}

// ---------------------------------------------------------------------------
// Per-step DNC state kernel: one workgroup per batch element (256 threads).
// ---------------------------------------------------------------------------
__global__ void __launch_bounds__(256)
k_state(int t,
        const float* __restrict__ gates, const int* __restrict__ src,
        const float* __restrict__ emb,
        const float* __restrict__ rk_w, const float* __restrict__ rk_b,
        const float* __restrict__ rs_w, const float* __restrict__ rs_b,
        const float* __restrict__ fg_w, const float* __restrict__ fg_b,
        const float* __restrict__ rm_w, const float* __restrict__ rm_b,
        const float* __restrict__ wk_w, const float* __restrict__ wk_b,
        const float* __restrict__ ws_w, const float* __restrict__ ws_b,
        const float* __restrict__ ev_w, const float* __restrict__ ev_b,
        const float* __restrict__ wv_w, const float* __restrict__ wv_b,
        const float* __restrict__ ag_w, const float* __restrict__ ag_b,
        const float* __restrict__ wg_w, const float* __restrict__ wg_b,
        float* s_g, float* u_g, float* p_g, float* ww_g, float* rw_g,
        float* mem_g, float* links_g,
        __bf16* hx, __bf16* xb) {
  const int b = blockIdx.x;
  const int tid = threadIdx.x;

  __shared__ float h_s[HIDD];
  __shared__ float mem_s[NSLOT * WDIM];
  __shared__ float iface[480];
  __shared__ float rw_o[NSLOT * HDN];   // old read weights, later reused as new
  __shared__ float f_s[NSLOT * HDN];
  __shared__ float b_s[NSLOT * HDN];
  __shared__ float cr_s[NSLOT * HDN];
  __shared__ float u_s[NSLOT], su_s[NSLOT], a_s[NSLOT], p_s[NSLOT];
  __shared__ float wwo[NSLOT], wwn[NSLOT], cw_s[NSLOT], sc_s[NSLOT];
  __shared__ float wkn[WDIM], evs[WDIM], wvs[WDIM];
  __shared__ float fgs[HDN], rknrm[HDN], betar[HDN], modes[HDN * 3];
  __shared__ float rv_s[WDIM * HDN];
  __shared__ float scal[8];   // 0:beta_w 1:ag 2:wg 3:wsum 4:mx 5:sm 6:knorm

  // ---- 1) LSTM pointwise + load carried state ----
  const float* g = gates + (long)b * G4;
  for (int i = tid; i < HIDD; i += 256) {
    float gi = g[i], gf = g[HIDD + i], gg = g[2 * HIDD + i], go = g[3 * HIDD + i];
    float sOld = s_g[(long)b * HIDD + i];
    float sNew = sigm(gf) * sOld + sigm(gi) * tanhf(gg);
    float hv = sigm(go) * tanhf(sNew);
    s_g[(long)b * HIDD + i] = sNew;
    h_s[i] = hv;
  }
  for (int i = tid; i < NSLOT * HDN; i += 256) rw_o[i] = rw_g[(long)b * NSLOT * HDN + i];
  if (tid < NSLOT) {
    u_s[tid] = u_g[(long)b * NSLOT + tid];
    p_s[tid] = p_g[(long)b * NSLOT + tid];
    wwo[tid] = ww_g[(long)b * NSLOT + tid];
  }
  for (int i = tid; i < NSLOT * WDIM; i += 256) mem_s[i] = mem_g[(long)b * NSLOT * WDIM + i];
  __syncthreads();

  // ---- 2) interface projections: 471 dot products of h with weight rows ----
  for (int id = tid; id < 471; id += 256) {
    const float* wr; float bias;
    if (id < 256)      { wr = rk_w + id * HIDD;          bias = rk_b[id]; }
    else if (id < 320) { int w = id - 256; wr = wk_w + w * HIDD; bias = wk_b[w]; }
    else if (id < 384) { int w = id - 320; wr = ev_w + w * HIDD; bias = ev_b[w]; }
    else if (id < 448) { int w = id - 384; wr = wv_w + w * HIDD; bias = wv_b[w]; }
    else if (id < 452) { int r = id - 448; wr = rs_w + r * HIDD; bias = rs_b[r]; }
    else if (id < 456) { int r = id - 452; wr = fg_w + r * HIDD; bias = fg_b[r]; }
    else if (id < 468) { int r = id - 456; wr = rm_w + r * HIDD; bias = rm_b[r]; }
    else if (id == 468){ wr = ws_w; bias = ws_b[0]; }
    else if (id == 469){ wr = ag_w; bias = ag_b[0]; }
    else               { wr = wg_w; bias = wg_b[0]; }
    float acc = bias;
    for (int k = 0; k < HIDD; ++k) acc += h_s[k] * wr[k];
    iface[id] = acc;
  }
  __syncthreads();

  // ---- 3) small nonlinearities ----
  if (tid < HDN) fgs[tid] = sigm(iface[452 + tid]);
  if (tid >= 64 && tid < 128) {
    int w = tid - 64;
    evs[w] = sigm(iface[320 + w]);
    wvs[w] = sigm(iface[384 + w]);
  }
  if (tid == 0) {
    scal[0] = oneplus_f(iface[468]);   // write strength
    scal[1] = sigm(iface[469]);        // allocation gate
    scal[2] = sigm(iface[470]);        // write gate
  }
  __syncthreads();

  // ---- 4) retention / usage update (uses carried ww) ----
  if (tid < NSLOT) {
    float psi = 1.0f;
#pragma unroll
    for (int hh = 0; hh < HDN; ++hh) psi *= (1.0f - fgs[hh] * rw_o[tid * HDN + hh]);
    float uo = u_s[tid], wo = wwo[tid];
    float un = (uo + wo - uo * wo) * psi;
    u_s[tid] = un;
    su_s[tid] = un;
  }
  __syncthreads();

  // ---- 5) bitonic sort (ascending) + allocation in sorted order ----
  for (int k = 2; k <= NSLOT; k <<= 1) {
    for (int j = k >> 1; j > 0; j >>= 1) {
      if (tid < NSLOT) {
        int ixj = tid ^ j;
        if (ixj > tid) {
          float x0 = su_s[tid], x1 = su_s[ixj];
          bool up = ((tid & k) == 0);
          if ((x0 > x1) == up) { su_s[tid] = x1; su_s[ixj] = x0; }
        }
      }
      __syncthreads();
    }
  }
  if (tid == 0) {
    float run = 1.0f;
    for (int i = 0; i < NSLOT; ++i) { a_s[i] = (1.0f - su_s[i]) * run; run *= su_s[i]; }
  }

  // ---- 6) write content attention (memory normalized in place, as reference) ----
  if (tid < NSLOT) {
    float ss = 0.0f; const int base = tid * WDIM;
    for (int j = 0; j < WDIM; ++j) { float v = mem_s[base + j]; ss += v * v; }
    float inv = 1.0f / (sqrtf(ss) + EPSV);
    for (int j = 0; j < WDIM; ++j) mem_s[base + j] *= inv;
  }
  if (tid == 132) {  // write-key norm (no EPS, matches reference)
    float ss = 0.0f;
    for (int j = 0; j < WDIM; ++j) { float v = iface[256 + j]; ss += v * v; }
    scal[6] = sqrtf(ss);
  }
  __syncthreads();
  if (tid < WDIM) wkn[tid] = iface[256 + tid] / scal[6];
  __syncthreads();
  if (tid < NSLOT) {
    float d = 0.0f; const int base = tid * WDIM;
    for (int j = 0; j < WDIM; ++j) d += mem_s[base + j] * wkn[j];
    sc_s[tid] = d * scal[0];
  }
  __syncthreads();
  if (tid == 0) { float mx = -1e30f; for (int i = 0; i < NSLOT; ++i) mx = fmaxf(mx, sc_s[i]); scal[4] = mx; }
  __syncthreads();
  if (tid < NSLOT) sc_s[tid] = expf(sc_s[tid] - scal[4]);
  __syncthreads();
  if (tid == 0) { float sm = 0.0f; for (int i = 0; i < NSLOT; ++i) sm += sc_s[i]; scal[5] = sm; }
  __syncthreads();
  if (tid < NSLOT) {
    cw_s[tid] = sc_s[tid] / scal[5];
    wwn[tid] = scal[2] * (scal[1] * a_s[tid] + (1.0f - scal[1]) * cw_s[tid]);
  }
  __syncthreads();

  // ---- 7) memory write, precedence prep, f/b zero ----
  for (int idx = tid; idx < NSLOT * WDIM; idx += 256) {
    int i = idx >> 6, j = idx & 63;
    mem_s[idx] += wwn[i] * (wvs[j] - evs[j]);
  }
  if (tid == 0) { float sm = 0.0f; for (int i = 0; i < NSLOT; ++i) sm += wwn[i]; scal[3] = sm; }
  for (int i = tid; i < NSLOT * HDN; i += 256) { f_s[i] = 0.0f; b_s[i] = 0.0f; }
  __syncthreads();
  if (tid < NSLOT) p_s[tid] = (1.0f - scal[3]) * p_s[tid] + wwn[tid];
  __syncthreads();

  // ---- 8) temporal link update (global RMW) + forward/backward accumulation ----
  float* L = links_g + (long)b * NSLOT * NSLOT;
  for (int idx = tid; idx < NSLOT * NSLOT; idx += 256) {
    int i = idx >> 7, j = idx & 127;
    float l = L[idx];
    l = l * (1.0f - wwn[i] - wwn[j]) + wwn[i] * p_s[j];
    if (i == j) l = 0.0f;
    L[idx] = l;
    if (l != 0.0f) {
#pragma unroll
      for (int hh = 0; hh < HDN; ++hh) {
        atomicAdd(&f_s[i * HDN + hh], l * rw_o[j * HDN + hh]);   // f = L @ rw
        atomicAdd(&b_s[j * HDN + hh], l * rw_o[i * HDN + hh]);   // b = L^T @ rw
      }
    }
  }
  __syncthreads();

  // ---- 9) read content attention (memory normalized again in place) ----
  if (tid < NSLOT) {
    float ss = 0.0f; const int base = tid * WDIM;
    for (int j = 0; j < WDIM; ++j) { float v = mem_s[base + j]; ss += v * v; }
    float inv = 1.0f / (sqrtf(ss) + EPSV);
    for (int j = 0; j < WDIM; ++j) mem_s[base + j] *= inv;
  }
  if (tid >= 128 && tid < 132) {
    int hh = tid - 128;
    float ss = 0.0f;
    for (int w = 0; w < WDIM; ++w) { float v = iface[hh * WDIM + w]; ss += v * v; }
    rknrm[hh] = sqrtf(ss);                    // no EPS (matches reference)
    betar[hh] = oneplus_f(iface[448 + hh]);
  }
  if (tid >= 132 && tid < 136) {              // read-mode softmax (over 3)
    int hh = tid - 132;
    float m0v = iface[456 + hh * 3], m1v = iface[456 + hh * 3 + 1], m2v = iface[456 + hh * 3 + 2];
    float mx = fmaxf(m0v, fmaxf(m1v, m2v));
    float e0 = expf(m0v - mx), e1 = expf(m1v - mx), e2 = expf(m2v - mx);
    float sm = e0 + e1 + e2;
    modes[hh * 3] = e0 / sm; modes[hh * 3 + 1] = e1 / sm; modes[hh * 3 + 2] = e2 / sm;
  }
  __syncthreads();
  for (int id = tid; id < NSLOT * HDN; id += 256) {
    int i = id & 127, hh = id >> 7;
    float d = 0.0f; const int base = i * WDIM;
    for (int j = 0; j < WDIM; ++j) d += mem_s[base + j] * iface[hh * WDIM + j];
    cr_s[i * HDN + hh] = d * (1.0f / rknrm[hh]) * betar[hh];
  }
  __syncthreads();
  if (tid < HDN) {
    int hh = tid;
    float mx = -1e30f;
    for (int i = 0; i < NSLOT; ++i) mx = fmaxf(mx, cr_s[i * HDN + hh]);
    float sm = 0.0f;
    for (int i = 0; i < NSLOT; ++i) { float e = expf(cr_s[i * HDN + hh] - mx); cr_s[i * HDN + hh] = e; sm += e; }
    float inv = 1.0f / sm;
    for (int i = 0; i < NSLOT; ++i) cr_s[i * HDN + hh] *= inv;
  }
  __syncthreads();

  // ---- 10) read weights + read vectors ----
  for (int id = tid; id < NSLOT * HDN; id += 256) {
    int hh = id & 3;
    float v = modes[hh * 3] * b_s[id] + modes[hh * 3 + 1] * cr_s[id] + modes[hh * 3 + 2] * f_s[id];
    rw_o[id] = v;                              // rw_o now holds new read weights
    rw_g[(long)b * NSLOT * HDN + id] = v;
  }
  __syncthreads();
  if (tid < WDIM * HDN) {
    int w = tid >> 2, hh = tid & 3;
    float acc = 0.0f;
    for (int i = 0; i < NSLOT; ++i) acc += mem_s[i * WDIM + w] * rw_o[i * HDN + hh];
    rv_s[tid] = acc;                           // layout [w][h] == reshape order w*HD+h
  }
  __syncthreads();

  // ---- 11) write back state and pack GEMM inputs ----
  for (int idx = tid; idx < NSLOT * WDIM; idx += 256) mem_g[(long)b * NSLOT * WDIM + idx] = mem_s[idx];
  if (tid < NSLOT) {
    u_g[(long)b * NSLOT + tid] = u_s[tid];
    p_g[(long)b * NSLOT + tid] = p_s[tid];
    ww_g[(long)b * NSLOT + tid] = wwn[tid];
  }
  __bf16* hxr = hx + (long)(t * BSZ + b) * KY;
  for (int i = tid; i < HIDD; i += 256) hxr[i] = (__bf16)h_s[i];
  if (tid < WDIM * HDN) hxr[HIDD + tid] = (__bf16)rv_s[tid];
  if (t < T_LEN - 1) {
    __bf16* xr = xb + (long)b * KG;
    int tok = src[(t + 1) * BSZ + b];
    const float* er = emb + (long)tok * EMB;
    for (int k = tid; k < EMB; k += 256) xr[k] = (__bf16)er[k];
    if (tid < WDIM * HDN) xr[EMB + tid] = (__bf16)rv_s[tid];
    for (int k = tid; k < HIDD; k += 256) xr[EMB + WDIM * HDN + k] = (__bf16)h_s[k];
  }
}

// ---------------------------------------------------------------------------
// Logits GEMM: out[2048][16000] = hx_bf[2048][1280] * Wy^T + Why_b
// 256 threads = 8 waves arranged 4(M)x2(N); wave tile 16x64 (4 accums).
// Grid (32, 125): WG tile 64x128, exact cover of 2048 x 16000.
// With TDM: A chunks (64x128 bf16) are double-buffered into LDS by the
// Tensor Data Mover (tensor_load_to_lds / s_wait_tensorcnt) while B streams
// directly from L2 into registers for the WMMAs.
// ---------------------------------------------------------------------------
#define KC  128               // K-chunk staged in LDS
#define NCH (KY / KC)         // 10 chunks

__global__ void __launch_bounds__(256)
k_logits(const __bf16* __restrict__ hx, const __bf16* __restrict__ Wy,
         const float* __restrict__ Why_b, float* __restrict__ out) {
  const int wave = threadIdx.x >> 5;
  const int lane = threadIdx.x & 31;
  const int hf = lane >> 4;
  const int lanelo = lane & 15;
  const int mrow = wave >> 1;                       // 0..3
  const int m0 = blockIdx.x * 64 + mrow * 16;
  const int n0 = blockIdx.y * 128 + (wave & 1) * 64;
  const __bf16* brow0 = Wy + (long)(n0 + lanelo) * KY;

  v8f acc[4];
#pragma unroll
  for (int s = 0; s < 4; ++s) acc[s] = (v8f){0.f, 0.f, 0.f, 0.f, 0.f, 0.f, 0.f, 0.f};

#if USE_TDM
  __shared__ __align__(16) __bf16 abuf[2][64 * KC];   // 2 x 16 KB
  const unsigned long gbase =
      (unsigned long)(const void*)(hx + (long)(blockIdx.x * 64) * KY);
  const int mloc = mrow * 16 + lanelo;                // 0..63 (row within WG tile)
  if (threadIdx.x < 32) {
    tdm_load_2d(gbase, (unsigned)(unsigned long)(void*)&abuf[0][0], KC, 64, KY);
  }
  for (int c = 0; c < NCH; ++c) {
    if (threadIdx.x < 32) __builtin_amdgcn_s_wait_tensorcnt(0);
    __syncthreads();                                  // chunk c visible to all waves
    if (c + 1 < NCH && threadIdx.x < 32) {
      tdm_load_2d(gbase + (unsigned long)(c + 1) * KC * 2,
                  (unsigned)(unsigned long)(void*)&abuf[(c + 1) & 1][0], KC, 64, KY);
    }
    const __bf16* arow = &abuf[c & 1][mloc * KC];
    for (int kk = 0; kk < KC; kk += 32) {
      const int k0 = c * KC + kk;
      __builtin_prefetch((const void*)(brow0 + k0 + 256), 0, 0);
      v8bf a0 = *(const v8bf*)(arow + kk + hf * 16);
      v8bf a1 = *(const v8bf*)(arow + kk + hf * 16 + 8);
      v16bf av;
#pragma unroll
      for (int j = 0; j < 8; ++j) { av[j] = a0[j]; av[j + 8] = a1[j]; }
#pragma unroll
      for (int s = 0; s < 4; ++s) {
        v16bf bv = *(const v16bf*)(brow0 + (long)s * 16 * KY + k0 + hf * 16);
        acc[s] = __builtin_amdgcn_wmma_f32_16x16x32_bf16(false, av, false, bv,
                                                         (short)0, acc[s], false, false);
      }
    }
    __syncthreads();                                  // done reading abuf[c&1]
  }
#else
  const int m = m0 + lanelo;
  const __bf16* arow = hx + (long)m * KY;
  for (int k0 = 0; k0 < KY; k0 += 32) {
    __builtin_prefetch((const void*)(brow0 + k0 + 128), 0, 0);
    v8bf a0 = *(const v8bf*)(arow + k0 + hf * 16);
    v8bf a1 = *(const v8bf*)(arow + k0 + hf * 16 + 8);
    v16bf av;
#pragma unroll
    for (int j = 0; j < 8; ++j) { av[j] = a0[j]; av[j + 8] = a1[j]; }
#pragma unroll
    for (int s = 0; s < 4; ++s) {
      v16bf bv = *(const v16bf*)(brow0 + (long)s * 16 * KY + k0 + hf * 16);
      acc[s] = __builtin_amdgcn_wmma_f32_16x16x32_bf16(false, av, false, bv,
                                                       (short)0, acc[s], false, false);
    }
  }
#endif

#pragma unroll
  for (int s = 0; s < 4; ++s) {
    int n = n0 + s * 16 + lanelo;
    float bb = Why_b[n];
#pragma unroll
    for (int r = 0; r < 8; ++r) {
      int mm = m0 + r + 8 * hf;
      out[(long)mm * OUTD + n] = acc[s][r] + bb;
    }
  }
}

// ---------------------------------------------------------------------------
// Host launch
// ---------------------------------------------------------------------------
extern "C" void kernel_launch(void* const* d_in, const int* in_sizes, int n_in,
                              void* d_out, int out_size, void* d_ws, size_t ws_size,
                              hipStream_t stream) {
  (void)in_sizes; (void)n_in; (void)out_size; (void)ws_size;

  const int*   src   = (const int*)d_in[0];
  const float* emb   = (const float*)d_in[1];
  const float* W_ih  = (const float*)d_in[2];
  const float* W_hh  = (const float*)d_in[3];
  const float* b_ih  = (const float*)d_in[4];
  const float* b_hh  = (const float*)d_in[5];
  const float* rk_w  = (const float*)d_in[6];
  const float* rk_b  = (const float*)d_in[7];
  const float* rs_w  = (const float*)d_in[8];
  const float* rs_b  = (const float*)d_in[9];
  const float* fg_w  = (const float*)d_in[10];
  const float* fg_b  = (const float*)d_in[11];
  const float* rm_w  = (const float*)d_in[12];
  const float* rm_b  = (const float*)d_in[13];
  const float* wk_w  = (const float*)d_in[14];
  const float* wk_b  = (const float*)d_in[15];
  const float* ws_w  = (const float*)d_in[16];
  const float* ws_b  = (const float*)d_in[17];
  const float* ev_w  = (const float*)d_in[18];
  const float* ev_b  = (const float*)d_in[19];
  const float* wv_w  = (const float*)d_in[20];
  const float* wv_b  = (const float*)d_in[21];
  const float* ag_w  = (const float*)d_in[22];
  const float* ag_b  = (const float*)d_in[23];
  const float* wg_w  = (const float*)d_in[24];
  const float* wg_b  = (const float*)d_in[25];
  const float* Why_w = (const float*)d_in[26];
  const float* Why_b = (const float*)d_in[27];
  const float* Wry_w = (const float*)d_in[28];

  char* p = (char*)d_ws;
  auto take = [&](size_t bytes) -> char* {
    char* r = p;
    p += (bytes + 255) & ~(size_t)255;
    return r;
  };
  __bf16* Wfull    = (__bf16*)take((size_t)G4 * KG * 2);          // 14.7 MB
  __bf16* Wy       = (__bf16*)take((size_t)OUTD * KY * 2);        // 41.0 MB
  float*  bias_sum = (float*) take((size_t)G4 * 4);
  __bf16* xb       = (__bf16*)take((size_t)BSZ * KG * 2);
  __bf16* hx       = (__bf16*)take((size_t)T_LEN * BSZ * KY * 2); // 5.2 MB
  float*  gates    = (float*) take((size_t)BSZ * G4 * 4);
  float*  s_g      = (float*) take((size_t)BSZ * HIDD * 4);
  float*  u_g      = (float*) take((size_t)BSZ * NSLOT * 4);
  float*  p_g      = (float*) take((size_t)BSZ * NSLOT * 4);
  float*  ww_g     = (float*) take((size_t)BSZ * NSLOT * 4);
  float*  rw_g     = (float*) take((size_t)BSZ * NSLOT * HDN * 4);
  float*  mem_g    = (float*) take((size_t)BSZ * NSLOT * WDIM * 4);
  float*  links_g  = (float*) take((size_t)BSZ * NSLOT * NSLOT * 4);

  k_convert<<<2048, 256, 0, stream>>>(W_ih, W_hh, b_ih, b_hh, Why_w, Wry_w,
                                      Wfull, Wy, bias_sum);
  k_init<<<512, 256, 0, stream>>>(src, emb, s_g, u_g, p_g, ww_g, rw_g,
                                  mem_g, links_g, xb);

  for (int t = 0; t < T_LEN; ++t) {
    k_gates<<<64, 256, 0, stream>>>(xb, Wfull, bias_sum, gates);
    k_state<<<BSZ, 256, 0, stream>>>(t, gates, src, emb,
                                     rk_w, rk_b, rs_w, rs_b, fg_w, fg_b,
                                     rm_w, rm_b, wk_w, wk_b, ws_w, ws_b,
                                     ev_w, ev_b, wv_w, wv_b, ag_w, ag_b,
                                     wg_w, wg_b,
                                     s_g, u_g, p_g, ww_g, rw_g, mem_g, links_g,
                                     hx, xb);
  }

  k_logits<<<dim3(32, 125), 256, 0, stream>>>(hx, Wy, Why_b, (float*)d_out);
}
